// Crossmamba_25847113187501
// MI455X (gfx1250) — matrix operand order repaired and checked by reading, not compiled, not run on hardware
//
#include <hip/hip_runtime.h>
#include <math.h>

// ============================================================================
// CrossMamba forward for MI455X (gfx1250, wave32, WMMA).
// GEMMs: bf16 WMMA (v_wmma_f32_16x16x32_bf16), f32 accumulate. 128x128 block
// tile, 8 waves, each wave 32x64 (8 WMMAs / K-tile). Tiles staged in LDS via
// float4 global loads + v_cvt_pk_bf16_f32 packed stores; B stored N-major so
// fragments are two contiguous ds_load_b128 each. OOB handled by index
// clamping (unconditional loads) -- garbage rows/cols only feed outputs the
// epilogue masks. K-chunks are uniform all-in/all-out (K % 8 == 0 always).
// ============================================================================

typedef __attribute__((ext_vector_type(16))) __bf16        v16bf;
typedef __attribute__((ext_vector_type(2)))  __bf16        v2bf;
typedef __attribute__((ext_vector_type(8)))  float         v8f;
typedef __attribute__((ext_vector_type(4)))  unsigned int  u32x4;
typedef __attribute__((ext_vector_type(2)))  unsigned int  u32x2;
struct U32x8 { u32x4 lo, hi; };

#define DM   512        // d_model
#define DIN  1024       // mamba inner dim = 2*d_model
#define DBCN 48         // dtr + 2*d_state
#define DTR  32

enum { ACT_NONE = 0, ACT_BIAS = 1, ACT_SILU = 2, ACT_GELU = 3, ACT_SOFTPLUS = 4 };

#define BM 128
#define BN 128
#define BK 32
#define ASTR 20   // dwords per LDS A row (16 data + 4 pad, keeps 16B alignment)
#define BSTR 20   // dwords per LDS B column (N-major)

__device__ __forceinline__ float apply_act(float v, int act) {
  if (act == ACT_SILU)          v = v / (1.0f + __expf(-v));
  else if (act == ACT_GELU)     v = 0.5f * v * (1.0f + erff(v * 0.70710678118f));
  else if (act == ACT_SOFTPLUS) v = (v > 20.0f) ? v : log1pf(expf(v));
  return v;
}

// pack two f32 -> one dword of bf16 pair; shaped to match v_cvt_pk_bf16_f32
__device__ __forceinline__ unsigned pack_bf2(float a, float b) {
  v2bf p; p[0] = (__bf16)a; p[1] = (__bf16)b;
  return __builtin_bit_cast(unsigned, p);
}

__global__ __launch_bounds__(256)
void gemm_wmma_k(const float* __restrict__ A, int lda,
                 const float* __restrict__ W,   // K x N row-major
                 const float* __restrict__ bias,
                 float* __restrict__ C,
                 int M, int N, int K, int act, int accum)
{
  __shared__ __attribute__((aligned(16))) unsigned sA32[BM][ASTR];
  __shared__ __attribute__((aligned(16))) unsigned sB32[BN][BSTR];   // [col][k/2]

  const int tid  = threadIdx.x;
  const int lane = tid & 31;
  const int wave = tid >> 5;
  const int wm   = wave >> 1;           // 0..3 -> 32-row group
  const int wn   = wave & 1;            // 0..1 -> 64-col group
  const int bm0  = blockIdx.y * BM;
  const int bn0  = blockIdx.x * BN;
  const int lrow = lane & 15;
  const int a0   = (lane < 16) ? 0 : 4; // A frag dword offset (K octet)
  const int b0   = (lane < 16) ? 0 : 4; // B frag dword offset (K octet)

  // staging decomposition (constant per thread)
  const int sc  = tid & 127;            // B: column within tile
  const int sko = (tid >> 7) * 2;       // B: first k-octet (0 or 2)

  v8f acc[2][4] = {};

  const int ktiles = (K + BK - 1) / BK;
  for (int kt = 0; kt < ktiles; ++kt) {
    const int k0 = kt * BK;

    // ---- stage A tile (128 x 32): float4 loads, packed bf16 b64 stores ----
#pragma unroll
    for (int it = 0; it < 4; ++it) {
      int idx = it * 256 + tid;
      int r = idx >> 3, q = idx & 7;            // row, float4-chunk
      int gr = bm0 + r; if (gr >= M) gr = M - 1;        // clamp: masked later
      float4 v = make_float4(0.f, 0.f, 0.f, 0.f);
      if (k0 + q * 4 < K)                               // K % 4 == 0 always
        v = *(const float4*)(A + (size_t)gr * lda + k0 + q * 4);
      u32x2 pk; pk[0] = pack_bf2(v.x, v.y); pk[1] = pack_bf2(v.z, v.w);
      *(u32x2*)&sA32[r][q * 2] = pk;
    }
    // ---- stage B tile (32 x 128) transposed to [col][k] ----
    {
      int gc = bn0 + sc; if (gc >= N) gc = N - 1;       // clamp: masked later
#pragma unroll
      for (int oo = 0; oo < 2; ++oo) {
        int oct = sko + oo;                             // k-octet 0..3
        u32x4 d = {0u, 0u, 0u, 0u};
        if (k0 + oct * 8 < K) {                         // K % 8 == 0 always
          const float* wp = W + (size_t)(k0 + oct * 8) * N + gc;
          float f0 = wp[0];
          float f1 = wp[(size_t)N];
          float f2 = wp[(size_t)2 * N];
          float f3 = wp[(size_t)3 * N];
          float f4 = wp[(size_t)4 * N];
          float f5 = wp[(size_t)5 * N];
          float f6 = wp[(size_t)6 * N];
          float f7 = wp[(size_t)7 * N];
          d[0] = pack_bf2(f0, f1); d[1] = pack_bf2(f2, f3);
          d[2] = pack_bf2(f4, f5); d[3] = pack_bf2(f6, f7);
        }
        *(u32x4*)&sB32[sc][oct * 4] = d;
      }
    }
    // ---- prefetch next K tile (global_prefetch_b8) ----
    if (kt + 1 < ktiles) {
      int pr = bm0 + (tid >> 1); if (pr >= M) pr = M - 1;
      int pc = k0 + BK; if (pc >= K) pc = K - 1;
      __builtin_prefetch(&A[(size_t)pr * lda + pc], 0, 1);
      int pk = k0 + BK + (tid >> 3); if (pk >= K) pk = K - 1;
      __builtin_prefetch(&W[(size_t)pk * N + bn0], 0, 1);
    }
    __syncthreads();

    // ---- fragments: two b128 LDS loads each, bit-cast to v16bf ----
    v16bf af[2], bfm[4];
#pragma unroll
    for (int mi = 0; mi < 2; ++mi) {
      int r = wm * 32 + mi * 16 + lrow;
      U32x8 t;
      t.lo = *(const u32x4*)&sA32[r][a0];
      t.hi = *(const u32x4*)&sA32[r][a0 + 8];
      af[mi] = __builtin_bit_cast(v16bf, t);
    }
#pragma unroll
    for (int ni = 0; ni < 4; ++ni) {
      int c = wn * 64 + ni * 16 + lrow;
      U32x8 t;
      t.lo = *(const u32x4*)&sB32[c][b0];
      t.hi = *(const u32x4*)&sB32[c][b0 + 8];
      bfm[ni] = __builtin_bit_cast(v16bf, t);
    }
#pragma unroll
    for (int mi = 0; mi < 2; ++mi)
#pragma unroll
      for (int ni = 0; ni < 4; ++ni)
        acc[mi][ni] = __builtin_amdgcn_wmma_f32_16x16x32_bf16(
            false, af[mi], false, bfm[ni], (short)0, acc[mi][ni], false, false);
    __syncthreads();
  }

  // ---- epilogue: bias + activation (+ accumulate), bounds-guarded ----
  const int rofs = (lane < 16) ? 0 : 8;
#pragma unroll
  for (int mi = 0; mi < 2; ++mi) {
#pragma unroll
    for (int ni = 0; ni < 4; ++ni) {
      int gcol = bn0 + wn * 64 + ni * 16 + lrow;
      if (gcol >= N) continue;
      float bv = bias ? bias[gcol] : 0.0f;
#pragma unroll
      for (int r = 0; r < 8; ++r) {
        int grow = bm0 + wm * 32 + mi * 16 + r + rofs;
        if (grow >= M) continue;
        float v = acc[mi][ni][r] + bv;
        v = apply_act(v, act);
        size_t o = (size_t)grow * N + gcol;
        if (accum) C[o] += v; else C[o] = v;
      }
    }
  }
}

// ---------------------------------------------------------------------------
// depthwise causal conv (K=4) over L, + SiLU.  input = u half of xz [T,2048]
// ---------------------------------------------------------------------------
__global__ __launch_bounds__(256)
void conv_silu_k(const float* __restrict__ xz, const float* __restrict__ cw,
                 const float* __restrict__ cb, float* __restrict__ uc,
                 int nseq, int L)
{
  int gid = blockIdx.x * 256 + threadIdx.x;
  int total = nseq * L * DIN;
  if (gid >= total) return;
  int c = gid & (DIN - 1);
  int tok = gid >> 10;
  int t = tok % L;
  int base = tok - t;
  float acc = cb[c];
#pragma unroll
  for (int k = 0; k < 4; ++k) {
    int tt = t + k - 3;
    if (tt >= 0) acc += xz[(size_t)(base + tt) * (2 * DIN) + c] * cw[c * 4 + k];
  }
  acc = acc / (1.0f + __expf(-acc));
  uc[(size_t)tok * DIN + c] = acc;
}

// ---------------------------------------------------------------------------
// selective scan, one thread per (seq, channel); fuses +uc*D and silu(z) gate.
// ---------------------------------------------------------------------------
__global__ __launch_bounds__(256)
void mamba_scan_k(const float* __restrict__ dt, const float* __restrict__ uc,
                  const float* __restrict__ dbc, const float* __restrict__ zbase,
                  const float* __restrict__ A_log, const float* __restrict__ Dp,
                  float* __restrict__ y, int nseq, int L)
{
  int gid = blockIdx.x * 256 + threadIdx.x;
  int d = gid & (DIN - 1);
  int seq = gid >> 10;
  if (seq >= nseq) return;
  float a[8], h[8];
#pragma unroll
  for (int n = 0; n < 8; ++n) { a[n] = -__expf(A_log[d * 8 + n]); h[n] = 0.0f; }
  const float Dd = Dp[d];
  for (int t = 0; t < L; ++t) {
    size_t tok = (size_t)seq * L + t;
    float dtv = dt[tok * DIN + d];
    float uv  = uc[tok * DIN + d];
    const float* bc = dbc + tok * DBCN;
    float du = dtv * uv;
    float yv = 0.0f;
#pragma unroll
    for (int n = 0; n < 8; ++n) {
      h[n] = h[n] * __expf(dtv * a[n]) + du * bc[DTR + n];
      yv  += h[n] * bc[DTR + 8 + n];
    }
    float zv = zbase[tok * (2 * DIN) + d];
    float sz = zv / (1.0f + __expf(-zv));
    y[tok * DIN + d] = (yv + uv * Dd) * sz;
  }
}

// ---------------------------------------------------------------------------
// out = rmsnorm(x + res) * w   (row = 512, one wave per row, 8 rows / block)
// res row index = row % res_mod (for broadcast positional residuals).
// ---------------------------------------------------------------------------
__global__ __launch_bounds__(256)
void add_rmsnorm_k(const float* __restrict__ x, const float* __restrict__ res,
                   int res_mod, const float* __restrict__ w,
                   float* __restrict__ out, int T)
{
  int row = blockIdx.x * 8 + (threadIdx.x >> 5);
  int lane = threadIdx.x & 31;
  if (row >= T) return;
  const float* xr = x + (size_t)row * DM;
  const float* rr = res ? res + (size_t)(row % res_mod) * DM : nullptr;
  float v[16]; float ss = 0.0f;
#pragma unroll
  for (int i = 0; i < 16; ++i) {
    float t = xr[lane + i * 32];
    if (rr) t += rr[lane + i * 32];
    v[i] = t; ss += t * t;
  }
#pragma unroll
  for (int off = 16; off > 0; off >>= 1) ss += __shfl_xor(ss, off, 32);
  float sc = rsqrtf(ss * (1.0f / DM) + 1e-6f);
#pragma unroll
  for (int i = 0; i < 16; ++i)
    out[(size_t)row * DM + lane + i * 32] = v[i] * sc * w[lane + i * 32];
}

// out = rmsnorm(a + tail(full)) * w ; full rows (nseq,Lfull,512), take skip..
__global__ __launch_bounds__(256)
void tail_add_rmsnorm_k(const float* __restrict__ a, const float* __restrict__ full,
                        int Lt, int Lfull, int skip, const float* __restrict__ w,
                        float* __restrict__ out, int T)
{
  int row = blockIdx.x * 8 + (threadIdx.x >> 5);
  int lane = threadIdx.x & 31;
  if (row >= T) return;
  int seq = row / Lt, t = row % Lt;
  const float* xr = a + (size_t)row * DM;
  const float* rr = full + (size_t)(seq * Lfull + skip + t) * DM;
  float v[16]; float ss = 0.0f;
#pragma unroll
  for (int i = 0; i < 16; ++i) {
    float tv = xr[lane + i * 32] + rr[lane + i * 32];
    v[i] = tv; ss += tv * tv;
  }
#pragma unroll
  for (int off = 16; off > 0; off >>= 1) ss += __shfl_xor(ss, off, 32);
  float sc = rsqrtf(ss * (1.0f / DM) + 1e-6f);
#pragma unroll
  for (int i = 0; i < 16; ++i)
    out[(size_t)row * DM + lane + i * 32] = v[i] * sc * w[lane + i * 32];
}

// (B,X,Y,512) -> (B,Y,X,512), float4 granularity
__global__ __launch_bounds__(256)
void permute_bxyd_k(const float4* __restrict__ in, float4* __restrict__ out,
                    int B, int X, int Y)
{
  int gid = blockIdx.x * 256 + threadIdx.x;
  int total = B * X * Y * 128;
  if (gid >= total) return;
  int c = gid & 127;
  int r = gid >> 7;
  int y = r % Y; int t = r / Y; int x = t % X; int b = t / X;
  out[((size_t)((b * Y + y) * X + x)) * 128 + c] = in[(size_t)r * 128 + c];
}

// concat along sequence: out (nseq, La+Lb, 512) from a (nseq,La,512), b (nseq,Lb,512)
__global__ __launch_bounds__(256)
void concat_seq_k(const float4* __restrict__ a, int La, const float4* __restrict__ b,
                  int Lb, float4* __restrict__ out, int nseq)
{
  int gid = blockIdx.x * 256 + threadIdx.x;
  int Lt = La + Lb;
  int total = nseq * Lt * 128;
  if (gid >= total) return;
  int c = gid & 127;
  int r = gid >> 7;
  int t = r % Lt; int s = r / Lt;
  out[gid] = (t < La) ? a[((size_t)s * La + t) * 128 + c]
                      : b[((size_t)s * Lb + (t - La)) * 128 + c];
}

// gather x_seq (16,1152,21) -> xs (16128,24) rows ordered (b, dim, cyc)
__global__ __launch_bounds__(256)
void gather_xseq_k(const float* __restrict__ xsq, float* __restrict__ xs)
{
  int gid = blockIdx.x * 256 + threadIdx.x;
  if (gid >= 16128 * 24) return;
  int t = gid % 24; int row = gid / 24;
  int ci = row % 48; int r2 = row / 48; int dd = r2 % 21; int b = r2 / 21;
  xs[gid] = xsq[((size_t)b * 1152 + ci * 24 + t) * 21 + dd];
}

// dst[i] = src[i % mod]
__global__ __launch_bounds__(256)
void bcast_rows_k(const float* __restrict__ src, float* __restrict__ dst,
                  int n, int mod)
{
  int gid = blockIdx.x * 256 + threadIdx.x;
  if (gid < n) dst[gid] = src[gid % mod];
}

// xn = a*x + om*N(0,1)  (hash-based Box-Muller; stand-in for JAX threefry)
__device__ __forceinline__ unsigned hash_u32(unsigned x) {
  x ^= x >> 16; x *= 0x7feb352du; x ^= x >> 15; x *= 0x846ca68bu; x ^= x >> 16;
  return x;
}
__global__ __launch_bounds__(256)
void noise_k(const float* __restrict__ x, float* __restrict__ out, int n,
             float a, float om, unsigned salt)
{
  int gid = blockIdx.x * 256 + threadIdx.x;
  if (gid >= n) return;
  unsigned h1 = hash_u32((unsigned)gid * 2u + 1u + salt * 0x9e3779b9u);
  unsigned h2 = hash_u32((unsigned)gid * 2u + 2u + salt * 0x85ebca6bu);
  float u1 = ((h1 >> 9) + 1u) * (1.0f / 8388609.0f);   // (0,1]
  float u2 = (h2 >> 9) * (1.0f / 8388608.0f);          // [0,1)
  float g = sqrtf(-2.0f * logf(u1)) * __cosf(6.28318530718f * u2);
  out[gid] = a * x[gid] + om * g;
}

// fp (16,21,24,24) -> out (16,576,21): out[b, c*24+t, d] = fp[b,d,c,t]
__global__ __launch_bounds__(256)
void final_permute_k(const float* __restrict__ fp, float* __restrict__ out)
{
  int gid = blockIdx.x * 256 + threadIdx.x;
  if (gid >= 16 * 576 * 21) return;
  int d = gid % 21; int r = gid / 21;
  int ct = r % 576; int b = r / 576;
  int c = ct / 24, t = ct % 24;
  out[gid] = fp[(((size_t)b * 21 + d) * 24 + c) * 24 + t];
}

// ============================================================================
// Host orchestration
// ============================================================================
struct MambaW { const float *in_proj,*conv_w,*conv_b,*x_proj,*dt_w,*dt_b,*A_log,*D,*out_proj; };
struct MlpW   { MambaW m; const float *w1,*b1,*w2,*b2,*n1,*n2; };
struct FdseW  { MlpW time, dim; };
struct DecW   { FdseW f; MambaW m; const float *w1,*b1,*w2,*b2,*n1,*n2,*pw,*pb; };

struct Arena {
  char* base; size_t cap; size_t off;
  float* alloc(size_t nfloats) {
    size_t bytes = (nfloats * 4 + 255) & ~(size_t)255;
    if (off + bytes > cap) off = 0;   // wrap (workspace sized by harness; ~1GB needed)
    float* p = (float*)(base + off); off += bytes; return p;
  }
};

static inline dim3 g1(long n) { return dim3((unsigned)((n + 255) / 256)); }
static inline dim3 grows(int T) { return dim3((unsigned)((T + 7) / 8)); }

static void gemm(hipStream_t s, const float* A, int lda, const float* W,
                 const float* bias, float* C, int M, int N, int K, int act, int accum)
{
  dim3 grid((N + BN - 1) / BN, (M + BM - 1) / BM);
  gemm_wmma_k<<<grid, 256, 0, s>>>(A, lda, W, bias, C, M, N, K, act, accum);
}

static void run_mamba(hipStream_t s, Arena& ar, const float* x, int nseq, int L,
                      const MambaW& w, float* out)
{
  const int T = nseq * L;
  size_t mark = ar.off;
  float* xz  = ar.alloc((size_t)T * 2 * DIN);
  float* uc  = ar.alloc((size_t)T * DIN);
  float* dbc = ar.alloc((size_t)T * DBCN);
  float* dt  = ar.alloc((size_t)T * DIN);
  float* y   = ar.alloc((size_t)T * DIN);
  gemm(s, x, DM, w.in_proj, nullptr, xz, T, 2 * DIN, DM, ACT_NONE, 0);
  conv_silu_k<<<g1((long)T * DIN), 256, 0, s>>>(xz, w.conv_w, w.conv_b, uc, nseq, L);
  gemm(s, uc, DIN, w.x_proj, nullptr, dbc, T, DBCN, DIN, ACT_NONE, 0);
  gemm(s, dbc, DBCN, w.dt_w, w.dt_b, dt, T, DIN, DTR, ACT_SOFTPLUS, 0);
  mamba_scan_k<<<g1((long)nseq * DIN), 256, 0, s>>>(dt, uc, dbc, xz + DIN,
                                                    w.A_log, w.D, y, nseq, L);
  gemm(s, y, DIN, w.out_proj, nullptr, out, T, DM, DIN, ACT_NONE, 0);
  ar.off = mark;
}

static void run_mammlp(hipStream_t s, Arena& ar, const float* x, int nseq, int L,
                       const MlpW& w, float* out)
{
  const int T = nseq * L;
  size_t mark = ar.off;
  float* z1 = ar.alloc((size_t)T * DM);
  size_t mark2 = ar.off;
  float* m = ar.alloc((size_t)T * DM);
  run_mamba(s, ar, x, nseq, L, w.m, m);
  add_rmsnorm_k<<<grows(T), 256, 0, s>>>(m, x, T, w.n1, z1, T);
  ar.off = mark2;
  float* h  = ar.alloc((size_t)T * 1024);
  float* h2 = ar.alloc((size_t)T * DM);
  gemm(s, z1, DM, w.w1, w.b1, h, T, 1024, DM, ACT_GELU, 0);
  gemm(s, h, 1024, w.w2, w.b2, h2, T, DM, 1024, ACT_BIAS, 0);
  add_rmsnorm_k<<<grows(T), 256, 0, s>>>(h2, z1, T, w.n1, out, T);
  ar.off = mark;
}

static void run_fdse(hipStream_t s, Arena& ar, const float* x, int b, int X, int Y,
                     const FdseW& w, float* out)
{
  const int T = b * X * Y;
  size_t mark = ar.off;
  float* t1 = ar.alloc((size_t)T * DM);
  run_mammlp(s, ar, x, b * X, Y, w.time, t1);
  float* t2 = ar.alloc((size_t)T * DM);
  permute_bxyd_k<<<g1((long)T * 128), 256, 0, s>>>((const float4*)t1, (float4*)t2, b, X, Y);
  float* t3 = ar.alloc((size_t)T * DM);
  run_mammlp(s, ar, t2, b * Y, X, w.dim, t3);
  permute_bxyd_k<<<g1((long)T * 128), 256, 0, s>>>((const float4*)t3, (float4*)out, b, Y, X);
  ar.off = mark;
}

static void run_declayer(hipStream_t s, Arena& ar, const float* x, const float* cross,
                         const DecW& w, float* fp, int accum, float* dec_out)
{
  const int T = 8064;           // 16*21*24 tokens
  size_t mark = ar.off;
  float* Zd = ar.alloc((size_t)T * DM);
  run_fdse(s, ar, x, 16, 21, 24, w.f, Zd);
  float* cat = ar.alloc((size_t)336 * 72 * DM);
  concat_seq_k<<<g1((long)336 * 72 * 128), 256, 0, s>>>(
      (const float4*)cross, 48, (const float4*)Zd, 24, (float4*)cat, 336);
  float* Zm = ar.alloc((size_t)336 * 72 * DM);
  run_mamba(s, ar, cat, 336, 72, w.m, Zm);
  float* Zt = ar.alloc((size_t)T * DM);
  tail_add_rmsnorm_k<<<grows(T), 256, 0, s>>>(Zd, Zm, 24, 72, 48, w.n1, Zt, T);
  float* h  = ar.alloc((size_t)T * DM);
  float* h2 = ar.alloc((size_t)T * DM);
  gemm(s, Zt, DM, w.w1, w.b1, h, T, DM, DM, ACT_GELU, 0);
  gemm(s, h, DM, w.w2, w.b2, h2, T, DM, DM, ACT_BIAS, 0);
  add_rmsnorm_k<<<grows(T), 256, 0, s>>>(h2, Zt, T, w.n2, dec_out, T);
  gemm(s, dec_out, DM, w.pw, w.pb, fp, T, 24, DM, ACT_BIAS, accum);
  ar.off = mark;
}

static void noise_scales(double stdv, float& a, float& om) {
  double ac = 1.0;
  for (int i = 0; i < 1000; ++i) { double x = stdv * (double)i / 999.0; ac *= (1.0 - x); }
  a = (float)sqrt(ac); om = (float)sqrt(1.0 - ac);
}

extern "C" void kernel_launch(void* const* d_in, const int* in_sizes, int n_in,
                              void* d_out, int out_size, void* d_ws, size_t ws_size,
                              hipStream_t stream)
{
  (void)in_sizes; (void)out_size;
  const float* x_seq = (const float*)d_in[0];
  int i = 1;
  auto nx = [&]() -> const float* { return (const float*)d_in[i++]; };
  auto getM = [&](MambaW& m) { m.in_proj=nx(); m.conv_w=nx(); m.conv_b=nx();
    m.x_proj=nx(); m.dt_w=nx(); m.dt_b=nx(); m.A_log=nx(); m.D=nx(); m.out_proj=nx(); };
  auto getMlp = [&](MlpW& w) { getM(w.m); w.w1=nx(); w.b1=nx(); w.w2=nx();
    w.b2=nx(); w.n1=nx(); w.n2=nx(); };
  auto getF = [&](FdseW& f) { getMlp(f.time); getMlp(f.dim); };
  auto getD = [&](DecW& d) { getF(d.f); getM(d.m); d.w1=nx(); d.b1=nx(); d.w2=nx();
    d.b2=nx(); d.n1=nx(); d.n2=nx(); d.pw=nx(); d.pb=nx(); };

  const float* emb_w = nx(); const float* emb_b = nx();
  const float* enc_pos = nx(); const float* dec_pos = nx();
  const float* pre_norm = nx();
  FdseW encW[4]; for (int k = 0; k < 4; ++k) getF(encW[k]);
  DecW decW[5];  for (int k = 0; k < 5; ++k) getD(decW[k]);
  if (n_in < i) return;   // unexpected flattening

  Arena ar{ (char*)d_ws, ws_size, 0 };

  // persistent buffers
  float* enc[5];
  for (int k = 0; k < 5; ++k) enc[k] = ar.alloc((size_t)16128 * DM);
  float* x0 = ar.alloc((size_t)8064 * DM);
  float* x1 = ar.alloc((size_t)8064 * DM);
  float* x2 = ar.alloc((size_t)8064 * DM);
  float* x3 = ar.alloc((size_t)8064 * DM);
  float* fp = ar.alloc((size_t)8064 * 24);

  // ---------------- embedding + pre-norm ----------------
  {
    size_t mk = ar.off;
    float* xs = ar.alloc((size_t)16128 * 24);
    gather_xseq_k<<<g1(16128L * 24), 256, 0, stream>>>(x_seq, xs);
    float* xe = ar.alloc((size_t)16128 * DM);
    gemm(stream, xs, 24, emb_w, emb_b, xe, 16128, DM, 24, ACT_BIAS, 0);
    add_rmsnorm_k<<<grows(16128), 256, 0, stream>>>(
        xe, enc_pos, 21 * 48, pre_norm, enc[0], 16128);
    ar.off = mk;
  }

  // ---------------- encoder: 4 FDSE blocks ----------------
  run_fdse(stream, ar, enc[0], 16, 21, 48, encW[0], enc[1]);
  const double stds[3] = { 0.2, 0.01, 0.001 };
  for (int k = 0; k < 3; ++k) {
    float a, om; noise_scales(stds[k], a, om);
    size_t mk = ar.off;
    float* xn = ar.alloc((size_t)16128 * DM);
    noise_k<<<g1(16128L * DM), 256, 0, stream>>>(enc[1], xn, 16128 * DM, a, om,
                                                 (unsigned)(k + 1));
    run_fdse(stream, ar, xn, 16, 21, 48, encW[k + 1], enc[2 + k]);
    ar.off = mk;
  }

  // ---------------- decoder: 5 layers ----------------
  bcast_rows_k<<<g1(16L * 21 * 24 * DM), 256, 0, stream>>>(
      dec_pos, x0, 16 * 21 * 24 * DM, 21 * 24 * DM);
  run_declayer(stream, ar, x0, enc[0], decW[0], fp, 0, x1);
  run_declayer(stream, ar, x1, enc[1], decW[1], fp, 1, x2);
  run_declayer(stream, ar, x2, enc[2], decW[2], fp, 1, x3);  // l2..l4 reuse x from l1
  run_declayer(stream, ar, x2, enc[3], decW[3], fp, 1, x3);
  run_declayer(stream, ar, x2, enc[4], decW[4], fp, 1, x3);

  final_permute_k<<<g1(16L * 576 * 21), 256, 0, stream>>>(fp, (float*)d_out);
}
